// BinaryLSTM_44976897523965
// MI455X (gfx1250) — matrix operand order repaired and checked by reading, not compiled, not run on hardware
//
#include <hip/hip_runtime.h>
#include <cstdint>
#include <cstddef>

#define BB 8192
#define DD 1024
#define HH 1024

typedef __attribute__((ext_vector_type(16))) __bf16 v16bf;
typedef __attribute__((ext_vector_type(8)))  float  v8f;

// ---------------- fp32 -> bf16 conversion (round to nearest even) ----------------
__device__ __forceinline__ unsigned short f2bf(float f) {
  unsigned int u = __float_as_uint(f);
  u += 0x7FFFu + ((u >> 16) & 1u);
  return (unsigned short)(u >> 16);
}

__global__ void cvt_f32_to_bf16(const float* __restrict__ in,
                                unsigned int* __restrict__ out, int n2) {
  int i = blockIdx.x * blockDim.x + threadIdx.x;
  if (i < n2) {
    float2 f = reinterpret_cast<const float2*>(in)[i];
    out[i] = (unsigned int)f2bf(f.x) | ((unsigned int)f2bf(f.y) << 16);
  }
}

// ---------------- WMMA fragment loaders ----------------
// A (16x32 bf16, 16-bit A layout): lane<16 -> K 0..7 & 16..23 ; lane>=16 -> K 8..15 & 24..31
__device__ __forceinline__ v16bf load_a(const unsigned short* __restrict__ base,
                                        int row, int kt, int lane) {
  int kb = kt + ((lane & 16) ? 8 : 0);
  const unsigned short* p = base + (size_t)row * DD + kb;
  union { uint4 u[2]; v16bf v; } f;
  f.u[0] = *reinterpret_cast<const uint4*>(p);
  f.u[1] = *reinterpret_cast<const uint4*>(p + 16);
  return f.v;
}

// B (32x16 bf16): lane holds column N=lane&15, K 0..7 & 16..23 (lane<16) / 8..15 & 24..31 (lane>=16)
__device__ __forceinline__ v16bf load_b(const unsigned short* __restrict__ tile, int lane) {
  const unsigned short* p = tile + (lane & 15) * 40 + ((lane & 16) ? 8 : 0);
  union { uint4 u[2]; v16bf v; } f;
  f.u[0] = *reinterpret_cast<const uint4*>(p);
  f.u[1] = *reinterpret_cast<const uint4*>(p + 16);
  return f.v;
}

__device__ __forceinline__ float sigm(float x) { return 1.0f / (1.0f + __expf(-x)); }

// ---------------- async staging: 14 x [16x32] bf16 weight tiles -> LDS ----------------
// 896 16-byte segments per chunk = 128 lanes x 7 -> waves 0..3 issue exactly 7 async ops each.
__device__ __forceinline__ void stage_async(const unsigned short* __restrict__ W,
                                            unsigned short* __restrict__ buf,
                                            int htile, int kt, int tid) {
  if (tid < 128) {
#pragma unroll
    for (int it = 0; it < 7; ++it) {
      int idx  = tid + it * 128;
      int gate = idx >> 6;
      int rem  = idx & 63;
      int row  = rem >> 2;
      int q    = rem & 3;
      const unsigned short* src =
          W + (size_t)gate * HH * DD + (size_t)(htile + row) * DD + kt + q * 8;
      // generic __shared__ pointer: low 32 bits are the LDS byte address
      unsigned lds = (unsigned)(size_t)(const void*)&buf[gate * (16 * 40) + row * 40 + q * 8];
      unsigned long long ga = (unsigned long long)(size_t)src;
      asm volatile("global_load_async_to_lds_b128 %0, %1, off"
                   :: "v"(lds), "v"(ga) : "memory");
    }
  }
}

// ---------------- fused tree-LSTM kernel ----------------
// gates: 0..3  use p  : Wd, Wf, Wo, Wi
//        4..8  use hl : Wdl, Wfll, Wfrl, Wol, Wil
//        9..13 use hr : Wdr, Wflr, Wfrr, Wor, Wir
__global__ __launch_bounds__(256, 1) void tree_lstm_wmma(
    const unsigned short* __restrict__ aP,
    const unsigned short* __restrict__ aL,
    const unsigned short* __restrict__ aR,
    const unsigned short* __restrict__ W,   // 14 * HH * DD bf16
    const float* __restrict__ clp, const float* __restrict__ crp,
    const float* __restrict__ bd,  const float* __restrict__ bfl,
    const float* __restrict__ bfr, const float* __restrict__ bo,
    const float* __restrict__ bi,
    float* __restrict__ hOut, float* __restrict__ cOut)
{
  constexpr int RS      = 40;        // padded row stride (ushorts): 80B -> conflict-free b128 reads
  constexpr int GATE_SZ = 16 * RS;   // 1280 B per gate tile
  constexpr int BUF_SZ  = 14 * GATE_SZ;
  __shared__ unsigned short smem[2 * BUF_SZ];   // 35.8 KB, double buffered

  const int tid   = threadIdx.x;
  const int lane  = tid & 31;
  const int wave  = tid >> 5;
  const int btile = blockIdx.x * 128 + wave * 16;
  const int htile = blockIdx.y * 16;
  const int m16   = lane & 15;
  const int arow  = btile + m16;

  v8f acc[14] = {};

  // prologue: stage chunk 0 into buffer 0; preload A fragments for chunk 0
  stage_async(W, smem, htile, 0, tid);
  v16bf fa_p = load_a(aP, arow, 0, lane);
  v16bf fa_l = load_a(aL, arow, 0, lane);
  v16bf fa_r = load_a(aR, arow, 0, lane);

  for (int kt = 0; kt < DD; kt += 32) {
    // own async stage of current chunk complete
    asm volatile("s_wait_asynccnt 0x0" ::: "memory");
    // all waves' stage done (current buffer valid) AND all waves finished
    // reading the other buffer last iteration (safe to overwrite it)
    __syncthreads();

    const unsigned short* buf  = smem + (((kt >> 5) & 1) ? BUF_SZ : 0);
    unsigned short*       nbuf = smem + (((kt >> 5) & 1) ? 0 : BUF_SZ);
    if (kt + 32 < DD) stage_async(W, nbuf, htile, kt + 32, tid);   // overlaps compute below

    // warm L2 two chunks ahead (global_prefetch_b8)
    if (kt + 64 < DD && tid < 14 * 16) {
      int gate = tid >> 4, row = tid & 15;
      __builtin_prefetch(W + (size_t)gate * HH * DD + (size_t)(htile + row) * DD + kt + 64, 0, 0);
    }

    // consume A fragments loaded last iteration; issue next-iteration A loads now
    // so their latency hides under the 14 WMMAs below
    v16bf ap = fa_p, al = fa_l, ar = fa_r;
    if (kt + 32 < DD) {
      fa_p = load_a(aP, arow, kt + 32, lane);
      fa_l = load_a(aL, arow, kt + 32, lane);
      fa_r = load_a(aR, arow, kt + 32, lane);
    }

    // two-deep rotating B-fragment pipeline: 4 ds_load_b128 outstanding at each
    // WMMA, so the scheduler can use partial s_wait_dscnt instead of full drains
    v16bf fb0 = load_b(buf, lane);
    v16bf fb1 = load_b(buf + GATE_SZ, lane);
#pragma unroll
    for (int g = 0; g < 14; g += 2) {
      v16bf a0 = (g < 4) ? ap : ((g < 9) ? al : ar);
      v16bf a1 = ((g + 1) < 4) ? ap : (((g + 1) < 9) ? al : ar);
      v16bf u0 = fb0, u1 = fb1;
      if (g + 2 < 14) fb0 = load_b(buf + (g + 2) * GATE_SZ, lane);
      if (g + 3 < 14) fb1 = load_b(buf + (g + 3) * GATE_SZ, lane);
      acc[g] = __builtin_amdgcn_wmma_f32_16x16x32_bf16(
          false, a0, false, u0, (short)0, acc[g], false, false);
      acc[g + 1] = __builtin_amdgcn_wmma_f32_16x16x32_bf16(
          false, a1, false, u1, (short)0, acc[g + 1], false, false);
    }
  }

  // epilogue: C/D layout -> lane gives N, VGPR r gives M (+8 for upper half-wave)
  const int   col  = htile + m16;
  const float vbd  = bd[col], vbfl = bfl[col], vbfr = bfr[col], vbo = bo[col], vbi = bi[col];
  const int   rowb = btile + ((lane & 16) ? 8 : 0);
#pragma unroll
  for (int r = 0; r < 8; ++r) {
    const size_t off = (size_t)(rowb + r) * HH + col;
    float i_g = sigm(acc[0][r] + acc[4][r]  + acc[9][r]  + vbd);
    float f_l = sigm(acc[1][r] + acc[5][r]  + acc[10][r] + vbfl);
    float f_r = sigm(acc[1][r] + acc[6][r]  + acc[11][r] + vbfr);
    float o_g = sigm(acc[2][r] + acc[7][r]  + acc[12][r] + vbo);
    float u_g = tanhf(acc[3][r] + acc[8][r] + acc[13][r] + vbi);
    float c   = i_g * u_g + f_l * clp[off] + f_r * crp[off];
    cOut[off] = c;
    hOut[off] = o_g * tanhf(c);
  }
}

// ---------------- host launcher ----------------
extern "C" void kernel_launch(void* const* d_in, const int* in_sizes, int n_in,
                              void* d_out, int out_size, void* d_ws, size_t ws_size,
                              hipStream_t stream) {
  // input order: 0 hl, 1 cl, 2 hr, 3 cr, 4 p, 5 Wd, 6 Wdl, 7 Wdr, 8 bd,
  //              9 Wf, 10 Wfll, 11 Wflr, 12 Wfrl, 13 Wfrr, 14 bfl, 15 bfr,
  //              16 Wo, 17 Wol, 18 Wor, 19 bo, 20 Wi, 21 Wil, 22 Wir, 23 bi
  const float* hl  = (const float*)d_in[0];
  const float* cl  = (const float*)d_in[1];
  const float* hr  = (const float*)d_in[2];
  const float* cr  = (const float*)d_in[3];
  const float* p   = (const float*)d_in[4];
  const float* bd  = (const float*)d_in[8];
  const float* bfl = (const float*)d_in[14];
  const float* bfr = (const float*)d_in[15];
  const float* bo  = (const float*)d_in[19];
  const float* bi  = (const float*)d_in[23];
  const int wIdx[14] = {5, 9, 16, 20,   6, 10, 12, 17, 21,   7, 11, 13, 18, 22};

  // ws layout (bf16): actP | actL | actR | 14 weight matrices  (~80 MB)
  unsigned short* ws   = (unsigned short*)d_ws;
  unsigned short* actP = ws;
  unsigned short* actL = ws + (size_t)BB * DD;
  unsigned short* actR = ws + 2 * (size_t)BB * DD;
  unsigned short* Wbf  = ws + 3 * (size_t)BB * DD;

  const int nAct2 = BB * DD / 2;
  cvt_f32_to_bf16<<<nAct2 / 256, 256, 0, stream>>>(p,  (unsigned int*)actP, nAct2);
  cvt_f32_to_bf16<<<nAct2 / 256, 256, 0, stream>>>(hl, (unsigned int*)actL, nAct2);
  cvt_f32_to_bf16<<<nAct2 / 256, 256, 0, stream>>>(hr, (unsigned int*)actR, nAct2);

  const int nW2 = HH * DD / 2;
  for (int g = 0; g < 14; ++g) {
    cvt_f32_to_bf16<<<nW2 / 256, 256, 0, stream>>>(
        (const float*)d_in[wIdx[g]], (unsigned int*)(Wbf + (size_t)g * HH * DD), nW2);
  }

  float* hOut = (float*)d_out;
  float* cOut = hOut + (size_t)BB * HH;
  dim3 grid(BB / 128, HH / 16);
  tree_lstm_wmma<<<grid, dim3(256), 0, stream>>>(actP, actL, actR, Wbf, cl, cr,
                                                 bd, bfl, bfr, bo, bi, hOut, cOut);
}
